// AttentionBlockDeberta_7035156430894
// MI455X (gfx1250) — matrix operand order repaired and checked by reading, not compiled, not run on hardware
//
#include <hip/hip_runtime.h>
#include <hip/hip_bf16.h>
#include <math.h>

typedef __attribute__((ext_vector_type(16))) _Float16 v16h;
typedef __attribute__((ext_vector_type(8)))  float    v8f;
typedef int v4i __attribute__((vector_size(16)));   // matches builtin param type

#define NH      12
#define DH      64
#define SEQ     1024
#define BATCH   4
#define HID     768
#define SPAN    256
#define TWOSPAN 512

#ifndef __has_builtin
#define __has_builtin(x) 0
#endif

#if __has_builtin(__builtin_amdgcn_global_load_async_to_lds_b128)
#define HAVE_ASYNC_COPY 1
#else
#define HAVE_ASYNC_COPY 0
#endif

#if HAVE_ASYNC_COPY
// One 16-byte async copy global->LDS (GLOBAL_LOAD_ASYNC_TO_LDS_B128, ASYNCcnt).
// Builtin signature (from this toolchain's diagnostic): arg0 is a non-const
// addrspace(1) v4i*, arg1 the LDS-side v4i*, then imm offset + imm cpol.
__device__ __forceinline__ void async_copy_b128(const void* g, void* l) {
  void* gnc = const_cast<void*>(g);
  __builtin_amdgcn_global_load_async_to_lds_b128(
      (__attribute__((address_space(1))) v4i*)gnc,
      (__attribute__((address_space(3))) v4i*)l,
      0, 0);
}
__device__ __forceinline__ void wait_async0() {
#if __has_builtin(__builtin_amdgcn_s_wait_asynccnt)
  __builtin_amdgcn_s_wait_asynccnt(0);
#else
  asm volatile("s_wait_asynccnt 0x0" ::: "memory");
#endif
}
#endif

union AF { v16h v; _Float16 h[16]; };
union CF { v8f  v; float    f[8];  };

__device__ __forceinline__ int iclampi(int x, int lo, int hi) {
  return x < lo ? lo : (x > hi ? hi : x);
}

// --- CDNA5 WMMA 16x16x32 f16 fragment index maps (ISA 7.12.2) ---
// A 16x32: lanes0-15 M=0..15 K={0..7,16..23}; lanes16-31 same M, K={8..15,24..31}
__device__ __forceinline__ int kpatA(int e, int half) {
  int j = e >> 1, p = e & 1;
  int base = (j < 4) ? (2 * j) : (16 + 2 * (j - 4));
  return base + (half ? 8 : 0) + p;
}
// B 32x16: lanes0-15 N=0..15 K={0..7,16..23}; lanes16-31 same N, K={8..15,24..31}
__device__ __forceinline__ int kpatB(int e, int half) {
  int j = e >> 1, p = e & 1;
  int base = (j < 4) ? (2 * j) : (16 + 2 * (j - 4));
  return base + (half ? 8 : 0) + p;
}

// A fragment from row-major [row][k] f16, row = mbase + (lane&15)
__device__ __forceinline__ v16h load_afrag_h(const _Float16* base, int row, int stride,
                                             int koff, int lane) {
  AF u; int half = lane >> 4;
#pragma unroll
  for (int e = 0; e < 16; ++e) u.h[e] = base[(size_t)row * stride + koff + kpatA(e, half)];
  return u.v;
}
// A fragment from row-major f32 (LDS score tile), converted to f16
__device__ __forceinline__ v16h load_afrag_f32(const float* base, int row, int stride,
                                               int koff, int lane) {
  AF u; int half = lane >> 4;
#pragma unroll
  for (int e = 0; e < 16; ++e)
    u.h[e] = (_Float16)base[(size_t)row * stride + koff + kpatA(e, half)];
  return u.v;
}
// B fragment where matrix is stored n-major [n][k] (i.e. computing X @ Wt)
__device__ __forceinline__ v16h load_bfrag_nmajor(const _Float16* base, int nrow, int stride,
                                                  int koff, int lane) {
  AF u; int half = lane >> 4;
#pragma unroll
  for (int e = 0; e < 16; ++e) u.h[e] = base[(size_t)nrow * stride + koff + kpatB(e, half)];
  return u.v;
}
// B fragment where matrix is stored k-major [k][n] (i.e. computing P @ V)
__device__ __forceinline__ v16h load_bfrag_kmajor(const _Float16* base, int col, int stride,
                                                  int koff, int lane) {
  AF u; int half = lane >> 4;
#pragma unroll
  for (int e = 0; e < 16; ++e) u.h[e] = base[(size_t)(koff + kpatB(e, half)) * stride + col];
  return u.v;
}

// DeBERTa log-bucket relative position (BUCKETS=256 -> mid=128, MAXPOS=512)
__device__ __forceinline__ int log_bucket(int rel) {
  int a = rel < 0 ? -rel : rel;
  if (a <= 128) return rel;            // abs_pos <= mid branch
  float lp = ceilf(__logf((float)a * (1.0f / 128.0f)) *
                   (127.0f / __logf(511.0f / 128.0f))) + 128.0f;
  int ip = (int)lp;
  return rel > 0 ? ip : -ip;
}

// ---------------------------------------------------------------------------
// Kernel 1: Y(f16) = X(f32, MxHID) @ W(f32, HIDxHID)^T + b, scattered into
// head-major (mode 0: [(b*NH+h)*SEQ+s][d]) or pos-major (mode 1: [h*2span+m][d]).
// Tile 64x64, 256 threads = 8 waves in 4x2 (16 rows x 32 cols per wave).
// ---------------------------------------------------------------------------
__global__ void proj_gemm_f32(const float* __restrict__ X, const float* __restrict__ W,
                              const float* __restrict__ bias, _Float16* __restrict__ Y,
                              int M, int mode) {
  __shared__ _Float16 As[64 * 32];
  __shared__ _Float16 Ws[64 * 32];
  const int tid = threadIdx.x, lane = tid & 31, w = tid >> 5;
  const int m0 = blockIdx.x * 64, n0 = blockIdx.y * 64;
  const int mbase = (w & 3) * 16, nbase = (w >> 2) * 32;
  CF acc0, acc1;
#pragma unroll
  for (int r = 0; r < 8; ++r) { acc0.f[r] = 0.f; acc1.f[r] = 0.f; }

  for (int k0 = 0; k0 < HID; k0 += 32) {
    __syncthreads();
#pragma unroll
    for (int i = 0; i < 8; ++i) {
      int idx = tid * 8 + i;                 // 0..2047
      int r = idx >> 5, c = idx & 31;
      As[r * 32 + c] = (_Float16)X[(size_t)(m0 + r) * HID + k0 + c];
      Ws[r * 32 + c] = (_Float16)W[(size_t)(n0 + r) * HID + k0 + c];
    }
    if (k0 + 32 < HID) {                     // global_prefetch_b8 for next K-slab
      __builtin_prefetch(&X[(size_t)(m0 + (tid >> 2)) * HID + k0 + 32], 0, 1);
      __builtin_prefetch(&W[(size_t)(n0 + (tid >> 2)) * HID + k0 + 32], 0, 1);
    }
    __syncthreads();
    v16h a  = load_afrag_h(As, mbase + (lane & 15), 32, 0, lane);
    v16h b0 = load_bfrag_nmajor(Ws, nbase + (lane & 15), 32, 0, lane);
    v16h b1 = load_bfrag_nmajor(Ws, nbase + 16 + (lane & 15), 32, 0, lane);
    acc0.v = __builtin_amdgcn_wmma_f32_16x16x32_f16(false, a, false, b0, (short)0, acc0.v, false, false);
    acc1.v = __builtin_amdgcn_wmma_f32_16x16x32_f16(false, a, false, b1, (short)0, acc1.v, false, false);
  }

  const int half = lane >> 4, col = lane & 15;
#pragma unroll
  for (int r = 0; r < 8; ++r) {
    int mg = m0 + mbase + r + 8 * half;
    if (mg >= M) continue;
#pragma unroll
    for (int nt = 0; nt < 2; ++nt) {
      int ng = n0 + nbase + nt * 16 + col;
      float v = (nt ? acc1.f[r] : acc0.f[r]) + bias[ng];
      int h = ng >> 6, dd = ng & 63;
      if (mode == 0) {
        int b = mg >> 10, s = mg & 1023;
        Y[(((size_t)b * NH + h) * SEQ + s) * DH + dd] = (_Float16)v;
      } else {
        Y[((size_t)h * TWOSPAN + mg) * DH + dd] = (_Float16)v;
      }
    }
  }
}

// ---------------------------------------------------------------------------
// Kernel 2: OUT(f16, per bh: SEQ x 512) = Ah(per bh: SEQ x 64) @ P(per h: 512 x 64)^T
// f16 output keeps both rel-attention tables (2 x 48 MB) resident in the 192MB L2,
// so the per-element gathers in the attention kernel do not touch HBM.
// Tile staging uses GLOBAL_LOAD_ASYNC_TO_LDS_B128 when the toolchain exposes it.
// ---------------------------------------------------------------------------
__global__ void relatt_gemm(const _Float16* __restrict__ Ah, const _Float16* __restrict__ P,
                            _Float16* __restrict__ OUT) {
  __shared__ __align__(16) _Float16 As[64 * 64];
  __shared__ __align__(16) _Float16 Ps[64 * 64];
  const int tid = threadIdx.x, lane = tid & 31, w = tid >> 5;
  const int bh = blockIdx.z, h = bh % NH;
  const int s0 = blockIdx.x * 64, n0 = blockIdx.y * 64;
  const _Float16* Abase = Ah + (size_t)bh * SEQ * DH;
  const _Float16* Pbase = P + (size_t)h * TWOSPAN * DH;
#if HAVE_ASYNC_COPY
  {
    // Tile is a contiguous 4096-half (8KB) region: (s0+r)*64 + c == s0*64 + idx
    int idx = tid * 16;
    const _Float16* ga = Abase + (size_t)s0 * DH + idx;
    const _Float16* gp = Pbase + (size_t)n0 * DH + idx;
    async_copy_b128(ga,     &As[idx]);
    async_copy_b128(ga + 8, &As[idx + 8]);
    async_copy_b128(gp,     &Ps[idx]);
    async_copy_b128(gp + 8, &Ps[idx + 8]);
    wait_async0();
  }
#else
#pragma unroll
  for (int i = 0; i < 16; ++i) {
    int idx = tid * 16 + i;                  // 0..4095
    int r = idx >> 6, c = idx & 63;
    As[idx] = Abase[(size_t)(s0 + r) * DH + c];
    Ps[idx] = Pbase[(size_t)(n0 + r) * DH + c];
  }
#endif
  __syncthreads();
  const int mbase = (w & 3) * 16, nbase = (w >> 2) * 32;
  CF acc0, acc1;
#pragma unroll
  for (int r = 0; r < 8; ++r) { acc0.f[r] = 0.f; acc1.f[r] = 0.f; }
#pragma unroll
  for (int kk = 0; kk < DH; kk += 32) {
    v16h a  = load_afrag_h(As, mbase + (lane & 15), 64, kk, lane);
    v16h b0 = load_bfrag_nmajor(Ps, nbase + (lane & 15), 64, kk, lane);
    v16h b1 = load_bfrag_nmajor(Ps, nbase + 16 + (lane & 15), 64, kk, lane);
    acc0.v = __builtin_amdgcn_wmma_f32_16x16x32_f16(false, a, false, b0, (short)0, acc0.v, false, false);
    acc1.v = __builtin_amdgcn_wmma_f32_16x16x32_f16(false, a, false, b1, (short)0, acc1.v, false, false);
  }
  const int half = lane >> 4, col = lane & 15;
#pragma unroll
  for (int r = 0; r < 8; ++r) {
    int mrow = mbase + r + 8 * half;
    size_t row = (size_t)bh * SEQ + s0 + mrow;
    OUT[row * TWOSPAN + n0 + nbase + col]      = (_Float16)acc0.f[r];
    OUT[row * TWOSPAN + n0 + nbase + 16 + col] = (_Float16)acc1.f[r];
  }
}

// ---------------------------------------------------------------------------
// Kernel 3: fused attention per (query-tile of 64, head, batch).
// Phase 1: scores = QK^T (WMMA) + c2p/p2c gathers (f16, L2-resident), -> 256KB LDS.
// Phase 2: row softmax in LDS; probs written to d_out (mandatory 201MB HBM write).
// Phase 3: ctx = P @ V (WMMA, split-K over 2 wave groups, LDS reduce) -> f16 ctx.
// Needs 288KB LDS -> only possible on CDNA5's 320KB/WGP.
// ---------------------------------------------------------------------------
__global__ void attn_kernel(const _Float16* __restrict__ Qh, const _Float16* __restrict__ Kh,
                            const _Float16* __restrict__ Vh, const _Float16* __restrict__ C2P,
                            const _Float16* __restrict__ P2C, float* __restrict__ probs_out,
                            _Float16* __restrict__ CtxH) {
  extern __shared__ char smem_raw[];
  float* sc  = (float*)smem_raw;                          // 64 x 1024 score tile
  float* red = (float*)(smem_raw + (size_t)64 * SEQ * 4); // 2 x (64x64) partial ctx
  const int tid = threadIdx.x, lane = tid & 31, w = tid >> 5;
  const int s0 = blockIdx.x * 64;
  const int h = blockIdx.y, b = blockIdx.z, bh = b * NH + h;
  const _Float16* Qb = Qh + (size_t)bh * SEQ * DH;
  const _Float16* Kb = Kh + (size_t)bh * SEQ * DH;
  const _Float16* Vb = Vh + (size_t)bh * SEQ * DH;
  const _Float16* c2p = C2P + (size_t)bh * SEQ * TWOSPAN;
  const _Float16* p2c = P2C + (size_t)bh * SEQ * TWOSPAN;
  const float inv_scale = rsqrtf((float)DH * 3.0f);

  // Preload Q fragments for this 64-row tile: 4 m-tiles x 2 K-steps over d=64
  v16h qf[4][2];
#pragma unroll
  for (int mt = 0; mt < 4; ++mt)
#pragma unroll
    for (int ks = 0; ks < 2; ++ks)
      qf[mt][ks] = load_afrag_h(Qb, s0 + mt * 16 + (lane & 15), DH, ks * 32, lane);

  // ---- Phase 1: scores tile ----
  for (int n0 = 0; n0 < SEQ; n0 += 128) {
    int c0 = n0 + w * 16;                    // 16 key-columns per wave
    v16h bf0 = load_bfrag_nmajor(Kb, c0 + (lane & 15), DH, 0, lane);
    v16h bf1 = load_bfrag_nmajor(Kb, c0 + (lane & 15), DH, 32, lane);
    CF acc[4];
#pragma unroll
    for (int mt = 0; mt < 4; ++mt) {
#pragma unroll
      for (int r = 0; r < 8; ++r) acc[mt].f[r] = 0.f;
      acc[mt].v = __builtin_amdgcn_wmma_f32_16x16x32_f16(false, qf[mt][0], false, bf0, (short)0, acc[mt].v, false, false);
      acc[mt].v = __builtin_amdgcn_wmma_f32_16x16x32_f16(false, qf[mt][1], false, bf1, (short)0, acc[mt].v, false, false);
    }
    const int half = lane >> 4, col = lane & 15;
#pragma unroll
    for (int mt = 0; mt < 4; ++mt) {
#pragma unroll
      for (int r = 0; r < 8; ++r) {
        int ml = mt * 16 + r + 8 * half;
        int sq = s0 + ml;
        int t  = c0 + col;
        int i1 = iclampi(log_bucket(sq - t) + SPAN, 0, TWOSPAN - 1);   // c2p idx
        int i2 = iclampi(-log_bucket(t - sq) + SPAN, 0, TWOSPAN - 1);  // p2c idx (post-swap)
        float v = acc[mt].f[r] + (float)c2p[(size_t)sq * TWOSPAN + i1]
                               + (float)p2c[(size_t)t  * TWOSPAN + i2];
        sc[(size_t)ml * SEQ + t] = v * inv_scale;
      }
    }
  }
  __syncthreads();

  // ---- Phase 2: softmax (8 rows per wave, 32-lane reductions) ----
  float* prow_base = probs_out + (size_t)bh * SEQ * SEQ;
#pragma unroll
  for (int rr = 0; rr < 8; ++rr) {
    int row = w * 8 + rr;
    float* srow = sc + (size_t)row * SEQ;
    float m = -3.4e38f;
    for (int t = lane; t < SEQ; t += 32) m = fmaxf(m, srow[t]);
#pragma unroll
    for (int off = 16; off > 0; off >>= 1) m = fmaxf(m, __shfl_xor(m, off, 32));
    float ssum = 0.f;
    for (int t = lane; t < SEQ; t += 32) {
      float e = __expf(srow[t] - m);
      srow[t] = e; ssum += e;
    }
#pragma unroll
    for (int off = 16; off > 0; off >>= 1) ssum += __shfl_xor(ssum, off, 32);
    float inv = 1.0f / ssum;
    float* prow = prow_base + (size_t)(s0 + row) * SEQ;
    for (int t = lane; t < SEQ; t += 32) {
      float p = srow[t] * inv;
      srow[t] = p;
      prow[t] = p;
    }
  }
  __syncthreads();

  // ---- Phase 3: ctx = P @ V, split-K across wave halves ----
  {
    const int mt = w & 3, khalf = w >> 2;
    CF acc[4];
#pragma unroll
    for (int nt = 0; nt < 4; ++nt)
#pragma unroll
      for (int r = 0; r < 8; ++r) acc[nt].f[r] = 0.f;
    for (int kk = khalf * 512; kk < khalf * 512 + 512; kk += 32) {
      v16h a = load_afrag_f32(sc, mt * 16 + (lane & 15), SEQ, kk, lane);
#pragma unroll
      for (int nt = 0; nt < 4; ++nt) {
        v16h bv = load_bfrag_kmajor(Vb, nt * 16 + (lane & 15), DH, kk, lane);
        acc[nt].v = __builtin_amdgcn_wmma_f32_16x16x32_f16(false, a, false, bv, (short)0, acc[nt].v, false, false);
      }
    }
    const int half = lane >> 4, col = lane & 15;
#pragma unroll
    for (int nt = 0; nt < 4; ++nt)
#pragma unroll
      for (int r = 0; r < 8; ++r) {
        int mrow = mt * 16 + r + 8 * half;
        red[khalf * 4096 + mrow * 64 + nt * 16 + col] = acc[nt].f[r];
      }
  }
  __syncthreads();
  for (int i = tid; i < 4096; i += 256) {
    float v = red[i] + red[4096 + i];
    int ml = i >> 6, dd = i & 63;
    CtxH[((size_t)b * SEQ + s0 + ml) * HID + h * DH + dd] = (_Float16)v;
  }
}

// ---------------------------------------------------------------------------
// Kernel 4: output projection + residual: Y(f32) = ctx(f16) @ Wo^T + bo + hidden
// ctx tile staging via async-to-LDS when available (no dtype conversion needed).
// ---------------------------------------------------------------------------
__global__ void out_proj_gemm(const _Float16* __restrict__ A, const float* __restrict__ W,
                              const float* __restrict__ bias, const float* __restrict__ resid,
                              float* __restrict__ Y) {
  __shared__ __align__(16) _Float16 As[64 * 32];
  __shared__ __align__(16) _Float16 Ws[64 * 32];
  const int tid = threadIdx.x, lane = tid & 31, w = tid >> 5;
  const int m0 = blockIdx.x * 64, n0 = blockIdx.y * 64;
  const int mbase = (w & 3) * 16, nbase = (w >> 2) * 32;
  CF acc0, acc1;
#pragma unroll
  for (int r = 0; r < 8; ++r) { acc0.f[r] = 0.f; acc1.f[r] = 0.f; }
  for (int k0 = 0; k0 < HID; k0 += 32) {
    __syncthreads();
    {
      int idx = tid * 8;
      int r = idx >> 5, c = idx & 31;
#if HAVE_ASYNC_COPY
      async_copy_b128(&A[(size_t)(m0 + r) * HID + k0 + c], &As[idx]);  // 8 halves
#else
#pragma unroll
      for (int i = 0; i < 8; ++i)
        As[idx + i] = A[(size_t)(m0 + r) * HID + k0 + c + i];
#endif
#pragma unroll
      for (int i = 0; i < 8; ++i)
        Ws[idx + i] = (_Float16)W[(size_t)(n0 + r) * HID + k0 + c + i];
    }
    if (k0 + 32 < HID) {
      __builtin_prefetch(&A[(size_t)(m0 + (tid >> 2)) * HID + k0 + 32], 0, 1);
      __builtin_prefetch(&W[(size_t)(n0 + (tid >> 2)) * HID + k0 + 32], 0, 1);
    }
#if HAVE_ASYNC_COPY
    wait_async0();
#endif
    __syncthreads();
    v16h a  = load_afrag_h(As, mbase + (lane & 15), 32, 0, lane);
    v16h b0 = load_bfrag_nmajor(Ws, nbase + (lane & 15), 32, 0, lane);
    v16h b1 = load_bfrag_nmajor(Ws, nbase + 16 + (lane & 15), 32, 0, lane);
    acc0.v = __builtin_amdgcn_wmma_f32_16x16x32_f16(false, a, false, b0, (short)0, acc0.v, false, false);
    acc1.v = __builtin_amdgcn_wmma_f32_16x16x32_f16(false, a, false, b1, (short)0, acc1.v, false, false);
  }
  const int half = lane >> 4, col = lane & 15;
#pragma unroll
  for (int r = 0; r < 8; ++r) {
    int mg = m0 + mbase + r + 8 * half;
#pragma unroll
    for (int nt = 0; nt < 2; ++nt) {
      int ng = n0 + nbase + nt * 16 + col;
      float v = (nt ? acc1.f[r] : acc0.f[r]) + bias[ng] + resid[(size_t)mg * HID + ng];
      Y[(size_t)mg * HID + ng] = v;
    }
  }
}

// ---------------------------------------------------------------------------
// Kernel 5: LayerNorm over last dim (768), one block per row.
// ---------------------------------------------------------------------------
__global__ void ln_kernel(const float* __restrict__ X, const float* __restrict__ g,
                          const float* __restrict__ bta, float* __restrict__ Y) {
  __shared__ float buf[256];
  const int row = blockIdx.x, tid = threadIdx.x;
  const float* xr = X + (size_t)row * HID;
  float s = 0.f;
  for (int i = tid; i < HID; i += 256) s += xr[i];
  buf[tid] = s; __syncthreads();
  for (int off = 128; off > 0; off >>= 1) {
    if (tid < off) buf[tid] += buf[tid + off];
    __syncthreads();
  }
  float mu = buf[0] / (float)HID;
  __syncthreads();
  float vv = 0.f;
  for (int i = tid; i < HID; i += 256) { float d = xr[i] - mu; vv += d * d; }
  buf[tid] = vv; __syncthreads();
  for (int off = 128; off > 0; off >>= 1) {
    if (tid < off) buf[tid] += buf[tid + off];
    __syncthreads();
  }
  float rstd = rsqrtf(buf[0] / (float)HID + 1e-7f);
  for (int i = tid; i < HID; i += 256)
    Y[(size_t)row * HID + i] = g[i] * (xr[i] - mu) * rstd + bta[i];
}

// ---------------------------------------------------------------------------
extern "C" void kernel_launch(void* const* d_in, const int* in_sizes, int n_in,
                              void* d_out, int out_size, void* d_ws, size_t ws_size,
                              hipStream_t stream) {
  (void)in_sizes; (void)n_in; (void)out_size; (void)ws_size;
  const float* hidden  = (const float*)d_in[0];
  const float* rel_emb = (const float*)d_in[1];
  const float* Wq = (const float*)d_in[2];
  const float* bq = (const float*)d_in[3];
  const float* Wk = (const float*)d_in[4];
  const float* bk = (const float*)d_in[5];
  const float* Wv = (const float*)d_in[6];
  const float* bv = (const float*)d_in[7];
  const float* Wo = (const float*)d_in[8];
  const float* bo = (const float*)d_in[9];
  const float* lw = (const float*)d_in[10];
  const float* lb = (const float*)d_in[11];

  float* out   = (float*)d_out;                                  // (4,1024,768)
  float* probs = out + (size_t)BATCH * SEQ * HID;                // (4,12,1024,1024)

  // Workspace carving (deterministic, no allocation)
  char* ws = (char*)d_ws;
  size_t off = 0;
  auto carve = [&](size_t bytes) -> char* {
    char* p = ws + off;
    off += (bytes + 255) & ~(size_t)255;
    return p;
  };
  _Float16* qh     = (_Float16*)carve((size_t)BATCH * NH * SEQ * DH * sizeof(_Float16));
  _Float16* kh     = (_Float16*)carve((size_t)BATCH * NH * SEQ * DH * sizeof(_Float16));
  _Float16* vh     = (_Float16*)carve((size_t)BATCH * NH * SEQ * DH * sizeof(_Float16));
  _Float16* posk   = (_Float16*)carve((size_t)NH * TWOSPAN * DH * sizeof(_Float16));
  _Float16* posq   = (_Float16*)carve((size_t)NH * TWOSPAN * DH * sizeof(_Float16));
  _Float16* c2p    = (_Float16*)carve((size_t)BATCH * NH * SEQ * TWOSPAN * sizeof(_Float16));
  _Float16* p2c    = (_Float16*)carve((size_t)BATCH * NH * SEQ * TWOSPAN * sizeof(_Float16));
  _Float16* ctxh   = (_Float16*)carve((size_t)BATCH * SEQ * HID * sizeof(_Float16));
  float*    outres = (float*)carve((size_t)BATCH * SEQ * HID * sizeof(float));

  dim3 blk(256);

  // Projections (X: 4096x768) -> head-major f16
  proj_gemm_f32<<<dim3(64, 12), blk, 0, stream>>>(hidden, Wq, bq, qh, BATCH * SEQ, 0);
  proj_gemm_f32<<<dim3(64, 12), blk, 0, stream>>>(hidden, Wk, bk, kh, BATCH * SEQ, 0);
  proj_gemm_f32<<<dim3(64, 12), blk, 0, stream>>>(hidden, Wv, bv, vh, BATCH * SEQ, 0);
  // Positional projections (512x768) -> per-head f16
  proj_gemm_f32<<<dim3(8, 12), blk, 0, stream>>>(rel_emb, Wk, bk, posk, TWOSPAN, 1);
  proj_gemm_f32<<<dim3(8, 12), blk, 0, stream>>>(rel_emb, Wq, bq, posq, TWOSPAN, 1);

  // Relative-attention tables (f16, L2-resident): c2p = Q @ pos_k^T, p2c = K @ pos_q^T
  relatt_gemm<<<dim3(16, 8, BATCH * NH), blk, 0, stream>>>(qh, posk, c2p);
  relatt_gemm<<<dim3(16, 8, BATCH * NH), blk, 0, stream>>>(kh, posq, p2c);

  // Fused attention: 288 KB dynamic LDS (CDNA5 320KB/WGP)
  int smem = 64 * SEQ * (int)sizeof(float) + 2 * 64 * 64 * (int)sizeof(float);
  (void)hipFuncSetAttribute((const void*)attn_kernel,
                            hipFuncAttributeMaxDynamicSharedMemorySize, smem);
  attn_kernel<<<dim3(SEQ / 64, NH, BATCH), blk, smem, stream>>>(qh, kh, vh, c2p, p2c,
                                                                probs, ctxh);

  // Output projection + residual, then LayerNorm
  out_proj_gemm<<<dim3(64, 12), blk, 0, stream>>>(ctxh, Wo, bo, hidden, outres);
  ln_kernel<<<dim3(BATCH * SEQ), blk, 0, stream>>>(outres, lw, lb, out);
}
